// HSAM_37546604102402
// MI455X (gfx1250) — matrix operand order repaired and checked by reading, not compiled, not run on hardware
//
#include <hip/hip_runtime.h>
#include <hip/hip_bf16.h>
#include <math.h>

// ---------------------------------------------------------------------------
// Trivial ext_vector types (union-safe, unlike HIP's constructor-laden uint4)
// ---------------------------------------------------------------------------
typedef __attribute__((ext_vector_type(16))) __bf16    v16bf;
typedef __attribute__((ext_vector_type(8)))  float     v8f;
typedef __attribute__((ext_vector_type(4)))  unsigned  u32x4;
typedef __attribute__((ext_vector_type(4)))  float     f32x4;

union FragBF { u32x4 q[2]; v16bf v; };
union Pack8  { unsigned short s[8]; u32x4 q; };

__device__ __forceinline__ unsigned short f2bf(float f) {
  unsigned u = __float_as_uint(f);
  u += 0x7FFFu + ((u >> 16) & 1u);             // round-to-nearest-even
  return (unsigned short)(u >> 16);
}

// tanh-GELU via sigmoid form (branch-free: v_exp_f32 + rcp, no libm tanh)
__device__ __forceinline__ float gelu_tanh(float x) {
  const float c2 = 1.5957691216057308f;        // 2*sqrt(2/pi)
  float u = c2 * (x + 0.044715f * x * x * x);
  return x / (1.0f + __expf(-u));
}

// float atomic max via signed/unsigned integer atomics (valid for IEEE754)
__device__ __forceinline__ void atomicMaxF(float* addr, float val) {
  if (val >= 0.0f) atomicMax((int*)addr, __float_as_int(val));
  else             atomicMin((unsigned int*)addr, __float_as_uint(val));
}

// ---------------------------------------------------------------------------
// fp32 -> bf16 bulk convert (8 elems/thread), optional fused GELU.
// n must be a multiple of 8 (always true here: N*128 / N*256).
// ---------------------------------------------------------------------------
__global__ void f32_to_bf16_kernel(const float* __restrict__ src,
                                   unsigned short* __restrict__ dst,
                                   long long n, int applyGelu)
{
  long long base = ((long long)blockIdx.x * blockDim.x + threadIdx.x) * 8;
  if (base >= n) return;
  f32x4 v0 = *(const f32x4*)(src + base);
  f32x4 v1 = *(const f32x4*)(src + base + 4);
  Pack8 p;
  #pragma unroll
  for (int i = 0; i < 4; ++i) {
    float a = v0[i], b = v1[i];
    if (applyGelu) { a = gelu_tanh(a); b = gelu_tanh(b); }
    p.s[i]     = f2bf(a);
    p.s[i + 4] = f2bf(b);
  }
  *(u32x4*)(dst + base) = p.q;
}

// W[K,128] fp32 -> Wt[128,K] bf16 (pre-transposed for the GEMM B operand)
__global__ void transpose_w_bf16(const float* __restrict__ W,
                                 unsigned short* __restrict__ Wt, int K)
{
  int t = blockIdx.x * blockDim.x + threadIdx.x;
  if (t >= 128 * K) return;
  int n = t / K, k = t - n * K;
  Wt[(size_t)n * K + k] = f2bf(W[(size_t)k * 128 + n]);
}

// ---------------------------------------------------------------------------
// Fold rel-einsum into projection weights, emitting bf16 TRANSPOSED weights:
//   WcT[j, i] = sum_d W[i, h*64+d] * rel[h,d,e]  (j = h*64+e; i==fi -> bias)
// ---------------------------------------------------------------------------
__global__ void compose_kernel(const float* __restrict__ W, const float* __restrict__ b,
                               const float* __restrict__ rel,
                               unsigned short* __restrict__ WcT, float* __restrict__ bc,
                               int fi)
{
  int t = blockIdx.x * blockDim.x + threadIdx.x;
  if (t >= (fi + 1) * 128) return;
  int i = t >> 7, j = t & 127, h = j >> 6, e = j & 63;
  const float* rw  = rel + (size_t)h * 4096;                 // [64][64]
  const float* src = (i == fi) ? b : (W + (size_t)i * 128);
  float acc = 0.0f;
  #pragma unroll 4
  for (int d = 0; d < 64; ++d) acc += src[h * 64 + d] * rw[d * 64 + e];
  if (i == fi) bc[j] = acc;
  else         WcT[(size_t)j * fi + i] = f2bf(acc);
}

// ---------------------------------------------------------------------------
// WMMA GEMM: Y[N,128] = A[N,K](bf16) @ Bt[128,K](bf16)^T + bias, fused epi.
// Block = 128 threads = 4 waves; tile M=64, N=128, K-step 32.
// Inner loop is pure b128 data movement + 8 x v_wmma_f32_16x16x32_bf16/wave.
// OOB rows are CLAMPED (not predicated) on load — garbage lanes are fine
// because the epilogue store is the only guarded operation.
// ---------------------------------------------------------------------------
#define TM 64
#define TN 128
#define TK 32
#define LDA 40   // padded LDS stride (ushorts), keeps 16B alignment
#define LDB 40

__global__ __launch_bounds__(128)
void wmma_gemm128(const unsigned short* __restrict__ A, int K, int Nrows,
                  const unsigned short* __restrict__ Bt, const float* __restrict__ bias,
                  const float* __restrict__ skipX, const float* __restrict__ skipGate,
                  int applyRelu, float* __restrict__ Y)
{
  __shared__ __align__(16) unsigned short As[TM][LDA];
  __shared__ __align__(16) unsigned short Bs[TN][LDB];

  const int tid  = threadIdx.x;
  const int lane = tid & 31;
  const int wave = tid >> 5;
  const int rowBase = blockIdx.x * TM;

  v8f acc[8] = {};

  const int aRow = tid >> 1;           // 64 rows, 2 threads/row
  const int aCol = (tid & 1) * 16;     // 16 consecutive bf16 each (32B)
  // clamp: always a valid address, no exec-mask juggling in the k-loop
  int ldRow = rowBase + aRow;
  if (ldRow >= Nrows) ldRow = Nrows - 1;
  const unsigned short* aBase = A + (size_t)ldRow * K + aCol;

  for (int k0 = 0; k0 < K; k0 += TK) {
    // ---- stage A tile (bf16, unconditional vectorized loads) ----
    {
      const unsigned short* src = aBase + k0;
      u32x4 a0 = *(const u32x4*)(src);
      u32x4 a1 = *(const u32x4*)(src + 8);
      __builtin_prefetch(src + TK, 0, 1);                    // global_prefetch_b8
      *(u32x4*)&As[aRow][aCol]     = a0;
      *(u32x4*)&As[aRow][aCol + 8] = a1;
    }
    // ---- stage B tile (bf16, pre-transposed in global: contiguous 64B/row) ----
    {
      const unsigned short* wsrc = Bt + (size_t)tid * K + k0;
      #pragma unroll
      for (int j = 0; j < 4; ++j)
        *(u32x4*)&Bs[tid][j * 8] = *(const u32x4*)(wsrc + j * 8);
    }
    __syncthreads();

    // ---- A fragment (16x32 bf16 per ISA layout) ----
    const int mloc = wave * 16 + (lane & 15);
    const int kA = (lane < 16) ? 0 : 8;              // lanes 0-15: K0..7,16..23
    FragBF a;
    a.q[0] = *(const u32x4*)&As[mloc][kA];
    a.q[1] = *(const u32x4*)&As[mloc][kA + 16];

    const int kB = (lane < 16) ? 0 : 8;              // lanes 0-15: K0..7,16..23
    #pragma unroll
    for (int t = 0; t < 8; ++t) {
      FragBF b;
      const unsigned short* bp = &Bs[t * 16 + (lane & 15)][kB];
      b.q[0] = *(const u32x4*)(bp);
      b.q[1] = *(const u32x4*)(bp + 16);
      acc[t] = __builtin_amdgcn_wmma_f32_16x16x32_bf16(
          false, a.v, false, b.v, (short)0, acc[t], false, false);
    }
    __syncthreads();
  }

  // ---- epilogue: bias (+sigmoid-gated skip-mix) (+relu), guarded store ----
  float sg = 0.0f;
  if (skipGate) sg = 1.0f / (1.0f + __expf(-skipGate[0]));
  #pragma unroll
  for (int t = 0; t < 8; ++t) {
    int col = t * 16 + (lane & 15);
    #pragma unroll
    for (int r = 0; r < 8; ++r) {
      int rloc = wave * 16 + ((lane < 16) ? r : (8 + r));
      int grow = rowBase + rloc;
      if (grow < Nrows) {
        float val = acc[t][r] + bias[col];
        if (skipX) val = sg * val + (1.0f - sg) * skipX[(size_t)grow * 128 + col];
        if (applyRelu) val = fmaxf(val, 0.0f);
        Y[(size_t)grow * 128 + col] = val;
      }
    }
  }
}

// ---------------------------------------------------------------------------
// Edge kernels (segment softmax over dst, per head). One wave32 per edge;
// lanes 0-15 = head 0 channels 0..63, lanes 16-31 = head 1 channels 64..127.
// ---------------------------------------------------------------------------
__global__ void edge_score_kernel(const float* __restrict__ q, const float* __restrict__ k,
                                  const int* __restrict__ src, const int* __restrict__ dst,
                                  const float* __restrict__ p_rel,
                                  float* __restrict__ alpha, float* __restrict__ nmax, int E)
{
  int e = blockIdx.x * 8 + (threadIdx.x >> 5);
  int lane = threadIdx.x & 31;
  if (e >= E) return;
  int s = src[e], d = dst[e];
  const float* qd = q + (size_t)d * 128;
  const float* ks = k + (size_t)s * 128;
  int base = lane * 4;
  f32x4 qv = *(const f32x4*)(qd + base);
  f32x4 kv = *(const f32x4*)(ks + base);
  float acc = qv[0] * kv[0] + qv[1] * kv[1] + qv[2] * kv[2] + qv[3] * kv[3];
  #pragma unroll
  for (int off = 8; off >= 1; off >>= 1) acc += __shfl_xor(acc, off, 32);
  int h = lane >> 4;
  if ((lane & 15) == 0) {
    float a = acc * p_rel[h] * 0.125f;               // 1/sqrt(64)
    alpha[(size_t)e * 2 + h] = a;
    atomicMaxF(&nmax[(size_t)d * 2 + h], a);
  }
}

__global__ void edge_norm_kernel(float* __restrict__ alpha, const float* __restrict__ nmax,
                                 float* __restrict__ nsum, const int* __restrict__ dst, int E)
{
  int idx = blockIdx.x * blockDim.x + threadIdx.x;
  if (idx >= 2 * E) return;
  int e = idx >> 1, h = idx & 1;
  int d = dst[e];
  float ex = __expf(alpha[idx] - nmax[(size_t)d * 2 + h]);
  alpha[idx] = ex;
  atomicAdd(&nsum[(size_t)d * 2 + h], ex);
}

__global__ void edge_agg_kernel(const float* __restrict__ v, const float* __restrict__ alpha,
                                const float* __restrict__ nsum,
                                const int* __restrict__ src, const int* __restrict__ dst,
                                float* __restrict__ agg, int E)
{
  int e = blockIdx.x * 8 + (threadIdx.x >> 5);
  int lane = threadIdx.x & 31;
  if (e >= E) return;
  int s = src[e], d = dst[e];
  int h = lane >> 4;
  float w = alpha[(size_t)e * 2 + h] / (nsum[(size_t)d * 2 + h] + 1e-16f);
  int c = lane * 4;
  f32x4 vv = *(const f32x4*)(v + (size_t)s * 128 + c);
  #pragma unroll
  for (int i = 0; i < 4; ++i)
    atomicAdd(&agg[(size_t)d * 128 + c + i], vv[i] * w);
}

// ---------------------------------------------------------------------------
// Pooling: node score w = relu(h@w1+b1)@w2+b2, segment softmax over batch,
// weighted segment sum into g[graph,128].
// ---------------------------------------------------------------------------
__global__ void node_score_kernel(const float* __restrict__ h,
                                  const float* __restrict__ w1, const float* __restrict__ wb1,
                                  const float* __restrict__ w2, const float* __restrict__ wb2,
                                  const int* __restrict__ batch,
                                  float* __restrict__ wnode, float* __restrict__ gmax, int N)
{
  int n = blockIdx.x * 8 + (threadIdx.x >> 5);
  int lane = threadIdx.x & 31;
  if (n >= N) return;
  const float* hn = h + (size_t)n * 128;
  float t = 0.0f;
  #pragma unroll 4
  for (int i = 0; i < 128; ++i) t += hn[i] * w1[i * 32 + lane];
  t = fmaxf(t + wb1[lane], 0.0f);
  float c = t * w2[lane];
  #pragma unroll
  for (int off = 16; off >= 1; off >>= 1) c += __shfl_xor(c, off, 32);
  if (lane == 0) {
    float w = c + wb2[0];
    wnode[n] = w;
    atomicMaxF(&gmax[batch[n]], w);
  }
}

__global__ void pool_norm_kernel(float* __restrict__ wnode, const float* __restrict__ gmax,
                                 float* __restrict__ gsum, const int* __restrict__ batch, int N)
{
  int n = blockIdx.x * blockDim.x + threadIdx.x;
  if (n >= N) return;
  int b = batch[n];
  float ex = __expf(wnode[n] - gmax[b]);
  wnode[n] = ex;
  atomicAdd(&gsum[b], ex);
}

__global__ void pool_agg_kernel(const float* __restrict__ h, const float* __restrict__ wnode,
                                const float* __restrict__ gsum, const int* __restrict__ batch,
                                float* __restrict__ g, int N)
{
  int n = blockIdx.x * 8 + (threadIdx.x >> 5);
  int lane = threadIdx.x & 31;
  if (n >= N) return;
  int b = batch[n];
  float coef = wnode[n] / (gsum[b] + 1e-16f);
  int c = lane * 4;
  f32x4 hv = *(const f32x4*)(h + (size_t)n * 128 + c);
  #pragma unroll
  for (int i = 0; i < 4; ++i)
    atomicAdd(&g[(size_t)b * 128 + c + i], hv[i] * coef);
}

// Final tiny MLP: out[g,256] = relu(g@m1+mb1)@m2+mb2. One block per graph.
__global__ void mlp_kernel(const float* __restrict__ g,
                           const float* __restrict__ m1, const float* __restrict__ mb1,
                           const float* __restrict__ m2, const float* __restrict__ mb2,
                           float* __restrict__ out)
{
  __shared__ float gl[128];
  __shared__ float tl[128];
  int gid = blockIdx.x, tid = threadIdx.x;
  if (tid < 128) gl[tid] = g[(size_t)gid * 128 + tid];
  __syncthreads();
  if (tid < 128) {
    float a = mb1[tid];
    #pragma unroll 4
    for (int i = 0; i < 128; ++i) a += gl[i] * m1[i * 128 + tid];
    tl[tid] = fmaxf(a, 0.0f);
  }
  __syncthreads();
  float o = mb2[tid];
  #pragma unroll 4
  for (int i = 0; i < 128; ++i) o += tl[i] * m2[i * 256 + tid];
  out[(size_t)gid * 256 + tid] = o;
}

__global__ void fill_kernel(float* __restrict__ p, float v, int n)
{
  int i = blockIdx.x * blockDim.x + threadIdx.x;
  if (i < n) p[i] = v;
}

// ---------------------------------------------------------------------------
// Host orchestration. Input order = jax pytree flatten (sorted dict keys):
//  0:x 1:edge_src 2:edge_dst 3:batch
//  4..15 conv1 / 16..27 conv2 / 28..39 conv3:
//     Wa,Wk,Wq,Wv,a_rel,ba,bk,bq,bv,m_rel,p_rel,skip
//  40:m1 41:m2 42:mb1 43:mb2 44:w1 45:w2 46:wb1 47:wb2 48:num_graphs
// ---------------------------------------------------------------------------
extern "C" void kernel_launch(void* const* d_in, const int* in_sizes, int n_in,
                              void* d_out, int out_size, void* d_ws, size_t ws_size,
                              hipStream_t stream) {
  const float* x    = (const float*)d_in[0];
  const int* esrc   = (const int*)d_in[1];
  const int* edst   = (const int*)d_in[2];
  const int* batch  = (const int*)d_in[3];
  const int N = in_sizes[0] / 256;
  const int E = in_sizes[1];
  const int G = out_size / 256;

  const float* m1  = (const float*)d_in[40];
  const float* m2  = (const float*)d_in[41];
  const float* mb1 = (const float*)d_in[42];
  const float* mb2 = (const float*)d_in[43];
  const float* w1  = (const float*)d_in[44];
  const float* w2  = (const float*)d_in[45];
  const float* wb1 = (const float*)d_in[46];
  const float* wb2 = (const float*)d_in[47];

  // carve workspace
  uintptr_t cur = (uintptr_t)d_ws;
  auto carve = [&](size_t bytes) -> void* {
    void* p = (void*)cur;
    cur += (bytes + 255) & ~(size_t)255;
    return p;
  };
  const size_t nf = (size_t)N * 128 * sizeof(float);
  float* qb    = (float*)carve(nf);
  float* kb    = (float*)carve(nf);
  float* vb    = (float*)carve(nf);
  float* agg   = (float*)carve(nf);
  float* hA    = (float*)carve(nf);
  float* hB    = (float*)carve(nf);
  unsigned short* Xbf   = (unsigned short*)carve((size_t)N * 256 * 2);  // bf16 activations
  unsigned short* aggbf = (unsigned short*)carve((size_t)N * 128 * 2);  // bf16 gelu(agg)
  unsigned short* WqT   = (unsigned short*)carve((size_t)128 * 256 * 2);
  unsigned short* WaT   = (unsigned short*)carve((size_t)128 * 128 * 2);
  unsigned short* WckT  = (unsigned short*)carve((size_t)128 * 256 * 2);
  unsigned short* WcvT  = (unsigned short*)carve((size_t)128 * 256 * 2);
  float* bck   = (float*)carve(128 * sizeof(float));
  float* bcv   = (float*)carve(128 * sizeof(float));
  float* alpha = (float*)carve((size_t)E * 2 * sizeof(float));
  float* nmax  = (float*)carve((size_t)N * 2 * sizeof(float));
  float* nsum  = (float*)carve((size_t)N * 2 * sizeof(float));
  float* wnode = (float*)carve((size_t)N * sizeof(float));
  float* gmax  = (float*)carve((size_t)G * sizeof(float));
  float* gsum  = (float*)carve((size_t)G * sizeof(float));
  float* gbuf  = (float*)carve((size_t)G * 128 * sizeof(float));
  (void)ws_size; (void)n_in;

  struct Layer { const float* X; int fi; const float* skip; float* out; int base; };
  Layer layers[3] = {
    { x,  256, nullptr, hA, 4  },
    { hA, 128, hA,      hB, 16 },
    { hB, 128, hB,      hA, 28 },
  };

  const dim3 gemmGrid((N + TM - 1) / TM);

  for (int L = 0; L < 3; ++L) {
    const int b = layers[L].base;
    const float* Wa    = (const float*)d_in[b + 0];
    const float* Wk    = (const float*)d_in[b + 1];
    const float* Wq    = (const float*)d_in[b + 2];
    const float* Wv    = (const float*)d_in[b + 3];
    const float* a_rel = (const float*)d_in[b + 4];
    const float* ba    = (const float*)d_in[b + 5];
    const float* bk    = (const float*)d_in[b + 6];
    const float* bq    = (const float*)d_in[b + 7];
    const float* bv    = (const float*)d_in[b + 8];
    const float* m_rel = (const float*)d_in[b + 9];
    const float* p_rel = (const float*)d_in[b + 10];
    const float* skipg = (const float*)d_in[b + 11];
    const float* X  = layers[L].X;
    const int fi    = layers[L].fi;
    const float* sk = layers[L].skip;
    float* out      = layers[L].out;

    // one-shot prep: bf16 activations + bf16 transposed (composed) weights
    long long nx = (long long)N * fi;
    f32_to_bf16_kernel<<<(unsigned)((nx / 8 + 255) / 256), 256, 0, stream>>>(X, Xbf, nx, 0);
    int ct = (fi + 1) * 128;
    compose_kernel<<<(ct + 255) / 256, 256, 0, stream>>>(Wk, bk, a_rel, WckT, bck, fi);
    compose_kernel<<<(ct + 255) / 256, 256, 0, stream>>>(Wv, bv, m_rel, WcvT, bcv, fi);
    transpose_w_bf16<<<(128 * fi + 255) / 256, 256, 0, stream>>>(Wq, WqT, fi);
    transpose_w_bf16<<<(128 * 128 + 255) / 256, 256, 0, stream>>>(Wa, WaT, 128);

    // node projections (WMMA, bf16 in / f32 accumulate)
    wmma_gemm128<<<gemmGrid, 128, 0, stream>>>(Xbf, fi, N, WqT,  bq,  nullptr, nullptr, 0, qb);
    wmma_gemm128<<<gemmGrid, 128, 0, stream>>>(Xbf, fi, N, WckT, bck, nullptr, nullptr, 0, kb);
    wmma_gemm128<<<gemmGrid, 128, 0, stream>>>(Xbf, fi, N, WcvT, bcv, nullptr, nullptr, 0, vb);

    // segment softmax + message aggregation over edges (bandwidth roofline)
    fill_kernel<<<(2 * N + 255) / 256, 256, 0, stream>>>(nmax, -INFINITY, 2 * N);
    hipMemsetAsync(nsum, 0, (size_t)N * 2 * sizeof(float), stream);
    hipMemsetAsync(agg, 0, nf, stream);
    edge_score_kernel<<<(E + 7) / 8, 256, 0, stream>>>(qb, kb, esrc, edst, p_rel, alpha, nmax, E);
    edge_norm_kernel<<<(2 * E + 255) / 256, 256, 0, stream>>>(alpha, nmax, nsum, edst, E);
    edge_agg_kernel<<<(E + 7) / 8, 256, 0, stream>>>(vb, alpha, nsum, esrc, edst, agg, E);

    // out = relu( mix( gelu(agg)@Wa + ba, skip ) )  — GELU fused into convert
    long long na = (long long)N * 128;
    f32_to_bf16_kernel<<<(unsigned)((na / 8 + 255) / 256), 256, 0, stream>>>(agg, aggbf, na, 1);
    wmma_gemm128<<<gemmGrid, 128, 0, stream>>>(aggbf, 128, N, WaT, ba, sk,
                                               sk ? skipg : nullptr, 1, out);
  }

  // attention-weighted pooling over batch (h = hA after layer 3)
  fill_kernel<<<(G + 255) / 256, 256, 0, stream>>>(gmax, -INFINITY, G);
  hipMemsetAsync(gsum, 0, (size_t)G * sizeof(float), stream);
  hipMemsetAsync(gbuf, 0, (size_t)G * 128 * sizeof(float), stream);
  node_score_kernel<<<(N + 7) / 8, 256, 0, stream>>>(hA, w1, wb1, w2, wb2, batch, wnode, gmax, N);
  pool_norm_kernel<<<(N + 255) / 256, 256, 0, stream>>>(wnode, gmax, gsum, batch, N);
  pool_agg_kernel<<<(N + 7) / 8, 256, 0, stream>>>(hA, wnode, gsum, batch, gbuf, N);

  // final MLP -> d_out [G,256]
  mlp_kernel<<<G, 256, 0, stream>>>(gbuf, m1, mb1, m2, mb2, (float*)d_out);
}
